// GeoReF_FeatNet_62414464745962
// MI455X (gfx1250) — compile-verified, matrix operand order checked by reading, not run on hardware
//
#include <hip/hip_runtime.h>

#define BN   4
#define NPTS 1024
#define KNB  10
#define SUPN 7

typedef __attribute__((ext_vector_type(16))) __bf16 v16bf;
typedef __attribute__((ext_vector_type(8)))  float  v8f;
typedef __attribute__((ext_vector_type(4)))  float  v4f;

// ---------------------------------------------------------------------------
// Generic batched strided GEMM:  C[b](m,n) = act( sum_k A[b](m,k)*B[b](n,k)
//                                              + bias[n] + res[b](m,n) )
// One wave computes a 16x32 tile: two v_wmma_f32_16x16x32_bf16 per k-step
// sharing one A fragment. AC/BC template flags select the contiguous-K fast
// path: each 16-element fragment = 4x global_load_b128 with immediate
// offsets (no per-element address math). K-tail is branchless clamp+select.
// ---------------------------------------------------------------------------
template <bool AC, bool BC>
__global__ void k_gemm_wmma(const float* __restrict__ A, const float* __restrict__ Bm,
                            float* __restrict__ C, const float* __restrict__ bias,
                            const float* __restrict__ res,
                            int M, int N, int K,
                            long sAb, long sAm, long sAk,
                            long sBb, long sBn, long sBk,
                            long sCb, long ldc, int act)
{
    const int lane = threadIdx.x & 31;
    const int m0 = blockIdx.x << 4;
    const int n0 = blockIdx.y << 5;
    const int b  = blockIdx.z;
    A  += (long)b * sAb;
    Bm += (long)b * sBb;
    C  += (long)b * sCb;
    const float* R = res ? res + (long)b * sCb : nullptr;

    const int half = lane >> 4, lm = lane & 15;
    // clamped row indices -> all loads in-bounds, no per-load predication
    const int am  = min(m0 + lm, M - 1);
    const int bn0 = min(n0 + lm, N - 1);
    const int bn1 = min(n0 + 16 + lm, N - 1);
    const float* __restrict__ Arow  = A  + (long)am  * sAm;
    const float* __restrict__ Brow0 = Bm + (long)bn0 * sBn;
    const float* __restrict__ Brow1 = Bm + (long)bn1 * sBn;

    v8f acc0 = {0.f, 0.f, 0.f, 0.f, 0.f, 0.f, 0.f, 0.f};
    v8f acc1 = {0.f, 0.f, 0.f, 0.f, 0.f, 0.f, 0.f, 0.f};

    const int kbase = half * 8;   // 16-bit A 16x32 layout (ISA 7.12.2)
    const int Kfull = K & ~31;

    int k0 = 0;
    for (; k0 < Kfull; k0 += 32) {
        v16bf af, bf0, bf1;
        if (AC) {
            const v4f* ap = reinterpret_cast<const v4f*>(Arow + k0 + kbase);
            const v4f q0 = ap[0], q1 = ap[1], q2 = ap[4], q3 = ap[5];
            __builtin_prefetch(Arow + k0 + kbase + 64, 0, 0);
#pragma unroll
            for (int e = 0; e < 4; ++e) {
                af[e]      = (__bf16)q0[e];
                af[4 + e]  = (__bf16)q1[e];
                af[8 + e]  = (__bf16)q2[e];
                af[12 + e] = (__bf16)q3[e];
            }
        } else {
#pragma unroll
            for (int p = 0; p < 8; ++p) {
                const int kk = k0 + kbase + (p < 4 ? (p << 1) : 16 + ((p - 4) << 1));
                af[2 * p]     = (__bf16)Arow[(long)kk * sAk];
                af[2 * p + 1] = (__bf16)Arow[(long)(kk + 1) * sAk];
            }
        }
        if (BC) {
            const v4f* bp0 = reinterpret_cast<const v4f*>(Brow0 + k0 + kbase);
            const v4f* bp1 = reinterpret_cast<const v4f*>(Brow1 + k0 + kbase);
            const v4f r0 = bp0[0], r1 = bp0[1], r2 = bp0[4], r3 = bp0[5];
            const v4f s0 = bp1[0], s1 = bp1[1], s2 = bp1[4], s3 = bp1[5];
            __builtin_prefetch(Brow0 + k0 + kbase + 64, 0, 0);
            __builtin_prefetch(Brow1 + k0 + kbase + 64, 0, 0);
#pragma unroll
            for (int e = 0; e < 4; ++e) {
                bf0[e]      = (__bf16)r0[e];
                bf0[4 + e]  = (__bf16)r1[e];
                bf0[8 + e]  = (__bf16)r2[e];
                bf0[12 + e] = (__bf16)r3[e];
                bf1[e]      = (__bf16)s0[e];
                bf1[4 + e]  = (__bf16)s1[e];
                bf1[8 + e]  = (__bf16)s2[e];
                bf1[12 + e] = (__bf16)s3[e];
            }
        } else {
#pragma unroll
            for (int p = 0; p < 8; ++p) {
                const int kk = k0 + kbase + (p < 4 ? (p << 1) : 16 + ((p - 4) << 1));
                bf0[2 * p]     = (__bf16)Brow0[(long)kk * sBk];
                bf0[2 * p + 1] = (__bf16)Brow0[(long)(kk + 1) * sBk];
                bf1[2 * p]     = (__bf16)Brow1[(long)kk * sBk];
                bf1[2 * p + 1] = (__bf16)Brow1[(long)(kk + 1) * sBk];
            }
        }
        acc0 = __builtin_amdgcn_wmma_f32_16x16x32_bf16(false, af, false, bf0,
                                                       (short)0, acc0, false, false);
        acc1 = __builtin_amdgcn_wmma_f32_16x16x32_bf16(false, af, false, bf1,
                                                       (short)0, acc1, false, false);
    }
    if (k0 < K) {   // K-tail: clamped loads + select-zero (branchless)
        v16bf af, bf0, bf1;
#pragma unroll
        for (int p = 0; p < 8; ++p) {
            const int kk = k0 + kbase + (p < 4 ? (p << 1) : 16 + ((p - 4) << 1));
            const long kc0 = min(kk, K - 1);
            const long kc1 = min(kk + 1, K - 1);
            float a0 = Arow [kc0 * sAk], a1 = Arow [kc1 * sAk];
            float b0 = Brow0[kc0 * sBk], b1 = Brow0[kc1 * sBk];
            float c0 = Brow1[kc0 * sBk], c1 = Brow1[kc1 * sBk];
            a0 = (kk     < K) ? a0 : 0.f;
            a1 = (kk + 1 < K) ? a1 : 0.f;
            b0 = (kk     < K) ? b0 : 0.f;
            b1 = (kk + 1 < K) ? b1 : 0.f;
            c0 = (kk     < K) ? c0 : 0.f;
            c1 = (kk + 1 < K) ? c1 : 0.f;
            af [2 * p] = (__bf16)a0; af [2 * p + 1] = (__bf16)a1;
            bf0[2 * p] = (__bf16)b0; bf0[2 * p + 1] = (__bf16)b1;
            bf1[2 * p] = (__bf16)c0; bf1[2 * p + 1] = (__bf16)c1;
        }
        acc0 = __builtin_amdgcn_wmma_f32_16x16x32_bf16(false, af, false, bf0,
                                                       (short)0, acc0, false, false);
        acc1 = __builtin_amdgcn_wmma_f32_16x16x32_bf16(false, af, false, bf1,
                                                       (short)0, acc1, false, false);
    }

    // store: C/D layout lanes0-15 -> M=r, lanes16-31 -> M=r+8; N = lane&15
#pragma unroll
    for (int t = 0; t < 2; ++t) {
        const int n = n0 + t * 16 + lm;
        if (n < N) {
            const float bv = bias ? bias[n] : 0.f;
            const v8f acc = t ? acc1 : acc0;
#pragma unroll
            for (int r = 0; r < 8; ++r) {
                const int m = m0 + half * 8 + r;
                if (m < M) {
                    float v = acc[r] + bv;
                    if (R)   v += R[(long)m * ldc + n];
                    if (act) v = fmaxf(v, 0.f);
                    C[(long)m * ldc + n] = v;
                }
            }
        }
    }
}

// ---------------------------------------------------------------------------
// KNN: one thread per point; keep 11 smallest squared distances (self incl.),
// emit indices 1..10 (drop self), matching top_k(-dist, k+1)[..., 1:].
// ---------------------------------------------------------------------------
__global__ void k_knn(const float* __restrict__ v, int* __restrict__ idx, int N)
{
    int n = blockIdx.x * blockDim.x + threadIdx.x;
    int b = blockIdx.y;
    if (n >= N) return;
    const float* vb = v + (long)b * N * 3;
    const float px = vb[n * 3 + 0], py = vb[n * 3 + 1], pz = vb[n * 3 + 2];
    float bd[KNB + 1];
    int   bi[KNB + 1];
#pragma unroll
    for (int t = 0; t <= KNB; ++t) { bd[t] = 3.4e38f; bi[t] = 0; }
    for (int j = 0; j < N; ++j) {
        const float dx = vb[j * 3 + 0] - px;
        const float dy = vb[j * 3 + 1] - py;
        const float dz = vb[j * 3 + 2] - pz;
        float d = dx * dx + dy * dy + dz * dz;
        int   jj = j;
#pragma unroll
        for (int t = 0; t <= KNB; ++t) {
            if (d < bd[t]) {
                float td = bd[t]; int ti = bi[t];
                bd[t] = d; bi[t] = jj;
                d = td; jj = ti;
            }
        }
    }
    int* ob = idx + ((long)b * N + n) * KNB;
#pragma unroll
    for (int k = 0; k < KNB; ++k) ob[k] = bi[k + 1];
}

// Column-normalize a (3, M) direction matrix.
__global__ void k_colnorm(const float* __restrict__ d, float* __restrict__ o, int M)
{
    int j = blockIdx.x * blockDim.x + threadIdx.x;
    if (j >= M) return;
    float a = d[j], b = d[M + j], c = d[2 * M + j];
    float nn = fmaxf(sqrtf(a * a + b * b + c * c), 1e-12f);
    o[j] = a / nn; o[M + j] = b / nn; o[2 * M + j] = c / nn;
}

// Fused hyperbolic-surface feature:
// out[b,n,c] = center + sum_s max_k relu(dirn(b,n,k) . dirs[:, s*C+c]) * support
// support = fout[b, idx, C + s*C + c] (or 1 if fout==null); center = fout[b,n,c] or 0.
__global__ void k_hs_feat(const float* __restrict__ vtx, const int* __restrict__ idx,
                          const float* __restrict__ dirs, const float* __restrict__ fout,
                          float* __restrict__ out, int N, int C)
{
    int c = threadIdx.x;
    int n = blockIdx.y;
    int b = blockIdx.z;
    if (c >= C) return;
    const float* vb = vtx + (long)b * N * 3;
    const int*   ib = idx + ((long)b * N + n) * KNB;
    const float px = vb[n * 3 + 0], py = vb[n * 3 + 1], pz = vb[n * 3 + 2];
    float dxk[KNB], dyk[KNB], dzk[KNB];
    int   jk[KNB];
#pragma unroll
    for (int k = 0; k < KNB; ++k) {
        const int j = ib[k];
        jk[k] = j;
        const float dx = vb[j * 3 + 0] - px;
        const float dy = vb[j * 3 + 1] - py;
        const float dz = vb[j * 3 + 2] - pz;
        const float l = fmaxf(sqrtf(dx * dx + dy * dy + dz * dz), 1e-12f);
        const float inv = 1.f / l;
        dxk[k] = dx * inv; dyk[k] = dy * inv; dzk[k] = dz * inv;
    }
    const int  SC = SUPN * C;
    const long rs = (long)(SUPN + 1) * C;
    const float* fb = fout ? fout + (long)b * N * rs : nullptr;
    const float center = fb ? fb[(long)n * rs + c] : 0.f;
    float accum = 0.f;
    for (int s = 0; s < SUPN; ++s) {
        const int m = s * C + c;
        const float d0 = dirs[m], d1 = dirs[SC + m], d2 = dirs[2 * SC + m];
        float best = -3.4e38f;
#pragma unroll
        for (int k = 0; k < KNB; ++k) {
            const float th  = fmaxf(dxk[k] * d0 + dyk[k] * d1 + dzk[k] * d2, 0.f);
            const float sup = fb ? fb[(long)jk[k] * rs + C + m] : 1.f;
            best = fmaxf(best, th * sup);
        }
        accum += best;
    }
    out[((long)b * N + n) * C + c] = center + accum;
}

// f_nb[b,n,c] = max_k feat[b, idx[b,n,k], c]
__global__ void k_gather_max(const float* __restrict__ f, const int* __restrict__ idx,
                             float* __restrict__ out, int N, int C)
{
    int c = threadIdx.x, n = blockIdx.y, b = blockIdx.z;
    if (c >= C) return;
    const int*   ib = idx + ((long)b * N + n) * KNB;
    const float* fb = f + (long)b * N * C;
    float m = -3.4e38f;
#pragma unroll
    for (int k = 0; k < KNB; ++k) m = fmaxf(m, fb[(long)ib[k] * C + c]);
    out[((long)b * N + n) * C + c] = m;
}

__global__ void k_mean_n(const float* __restrict__ x, float* __restrict__ g, int N, int C)
{
    int i = blockIdx.x * blockDim.x + threadIdx.x;
    if (i >= BN * C) return;
    int b = i / C, c = i % C;
    const float* xb = x + (long)b * N * C + c;
    float s = 0.f;
    for (int n = 0; n < N; ++n) s += xb[(long)n * C];
    g[i] = s / (float)N;
}

__global__ void k_rowmax(const float* __restrict__ x, float* __restrict__ g, int N, int C)
{
    int i = blockIdx.x * blockDim.x + threadIdx.x;
    if (i >= BN * C) return;
    int b = i / C, c = i % C;
    const float* xb = x + (long)b * N * C + c;
    float m = -3.4e38f;
    for (int n = 0; n < N; ++n) m = fmaxf(m, xb[(long)n * C]);
    g[i] = m;
}

__global__ void k_bcast_add(float* __restrict__ out, const float* __restrict__ g,
                            long total, int N, int C)
{
    long i = blockIdx.x * (long)blockDim.x + threadIdx.x;
    if (i >= total) return;
    int c = (int)(i % C);
    int b = (int)(i / ((long)N * C));
    out[i] += g[b * C + c];
}

__global__ void k_add(float* __restrict__ out, const float* __restrict__ a,
                      const float* __restrict__ b2, long total, int relu)
{
    long i = blockIdx.x * (long)blockDim.x + threadIdx.x;
    if (i >= total) return;
    float v = a[i] + b2[i];
    if (relu) v = fmaxf(v, 0.f);
    out[i] = v;
}

// dst (B,C,N) = relu( a(B,N,C) + b2(B,N,C) ) transposed
__global__ void k_add_tr_relu(float* __restrict__ dst, const float* __restrict__ a,
                              const float* __restrict__ b2, int N, int C, long total)
{
    long i = blockIdx.x * (long)blockDim.x + threadIdx.x;
    if (i >= total) return;
    int  c = (int)(i % C);
    long t = i / C;
    int  n = (int)(t % N);
    int  b = (int)(t / N);
    dst[((long)b * C + c) * N + n] = fmaxf(a[i] + b2[i], 0.f);
}

// dst (B,N,C) = xr(B,C,N) + xts(B,C,N)  (sum + transpose)
__global__ void k_sum_tr(float* __restrict__ dst, const float* __restrict__ xr,
                         const float* __restrict__ xts, int N, int C, long total)
{
    long i = blockIdx.x * (long)blockDim.x + threadIdx.x;
    if (i >= total) return;
    int  c = (int)(i % C);
    long t = i / C;
    int  n = (int)(t % N);
    int  b = (int)(t / N);
    long si = ((long)b * C + c) * N + n;
    dst[i] = xr[si] + xts[si];
}

__global__ void k_add_eye(float* __restrict__ g, int dim)
{
    int i = blockIdx.x * blockDim.x + threadIdx.x;
    if (i >= BN * dim) return;
    int b = i / dim, d = i % dim;
    g[(long)b * dim * dim + (long)d * dim + d] += 1.f;
}

// out (B,1088,N): rows 0..1023 = g[b,r] broadcast; rows 1024..1087 = x(B,64,N)
__global__ void k_concat(float* __restrict__ out, const float* __restrict__ g,
                         const float* __restrict__ x, int N, long total)
{
    long i = blockIdx.x * (long)blockDim.x + threadIdx.x;
    if (i >= total) return;
    int  n = (int)(i % N);
    long t = i / N;
    int  r = (int)(t % 1088);
    int  b = (int)(t / 1088);
    out[i] = (r < 1024) ? g[b * 1024 + r]
                        : x[((long)b * 64 + (r - 1024)) * N + n];
}

// dst (B,N,3) = v(B,3,N) transposed
__global__ void k_tr3(float* __restrict__ dst, const float* __restrict__ v, int N, long total)
{
    long i = blockIdx.x * (long)blockDim.x + threadIdx.x;
    if (i >= total) return;
    int  c = (int)(i % 3);
    long t = i / 3;
    int  n = (int)(t % N);
    int  b = (int)(t / N);
    dst[i] = v[((long)b * 3 + c) * N + n];
}

// ---------------------------------------------------------------------------
extern "C" void kernel_launch(void* const* d_in, const int* in_sizes, int n_in,
                              void* d_out, int out_size, void* d_ws, size_t ws_size,
                              hipStream_t stream)
{
    (void)in_sizes; (void)n_in; (void)out_size; (void)ws_size;
    const int N = NPTS, B = BN;

    const float* v_pcl   = (const float*)d_in[0];
    const float* v_prior = (const float*)d_in[1];
    int pi = 2;
    auto nextp = [&]() { return (const float*)d_in[pi++]; };
    const float* pm_c1_w  = nextp(); const float* pm_c1_b  = nextp();
    const float* pm_c2_w  = nextp(); const float* pm_c2_b  = nextp();
    const float* pm_c3_w  = nextp(); const float* pm_c3_b  = nextp();
    const float* pm_fc1_w = nextp(); const float* pm_fc1_b = nextp();
    const float* pm_fc2_w = nextp(); const float* pm_fc2_b = nextp();
    const float* pm_fc3_w = nextp(); const float* pm_fc3_b = nextp();
    const float* hs1_dirs = nextp();
    const float* hs1_ste_w = nextp(); const float* hs1_ste_b = nextp();
    const float* hs1_orl_w = nextp(); const float* hs1_orl_b = nextp();
    const float* hs2_W = nextp(); const float* hs2_b = nextp();
    const float* hs2_dirs = nextp();
    const float* hs2_ste_w = nextp(); const float* hs2_ste_b = nextp();
    const float* hs2_orl_w = nextp(); const float* hs2_orl_b = nextp();
    const float* c3_w = nextp(); const float* c3_b = nextp();
    const float* c4_w = nextp(); const float* c4_b = nextp();
    const float* fm1_W = nextp(); const float* fm1_b = nextp();
    const float* fm1_dirs = nextp();
    const float* fm1_ste_w = nextp(); const float* fm1_ste_b = nextp();
    const float* fm1_orl_w = nextp(); const float* fm1_orl_b = nextp();
    const float* fm2_W = nextp(); const float* fm2_b = nextp();
    const float* fm2_dirs = nextp();
    const float* fm2_ste_w = nextp(); const float* fm2_ste_b = nextp();
    const float* fm2_orl_w = nextp(); const float* fm2_orl_b = nextp();
    const float* fm_c3_w  = nextp(); const float* fm_c3_b  = nextp();
    const float* fm_fc1_w = nextp(); const float* fm_fc1_b = nextp();
    const float* fm_fc2_w = nextp(); const float* fm_fc2_b = nextp();
    const float* fm_fc3_w = nextp(); const float* fm_fc3_b = nextp();
    const float* fm_fc4_w = nextp(); const float* fm_fc4_b = nextp();

    // ---- workspace bump allocator ----
    char*  wsb = (char*)d_ws;
    size_t off = 0;
    auto allocf = [&](size_t n) { float* p = (float*)(wsb + off); off += ((n * 4 + 255) / 256) * 256; return p; };
    auto alloci = [&](size_t n) { int*   p = (int*)(wsb + off);   off += ((n * 4 + 255) / 256) * 256; return p; };

    float* dcn_hs1 = allocf(3 * 448);
    float* dcn_fm1 = allocf(3 * 448);
    float* dcn_fm2 = allocf(3 * 896);
    float* dcn_hs2 = allocf(3 * 896);
    float* xr_pcl  = allocf((size_t)B * 64 * N);
    float* xts_pcl = allocf((size_t)B * 64 * N);
    float* xr_p    = allocf((size_t)B * 64 * N);
    float* xts_p   = allocf((size_t)B * 64 * N);
    float* xr2     = allocf((size_t)B * 64 * N);
    float* xts2    = allocf((size_t)B * 64 * N);
    float* pt_r    = allocf((size_t)B * 4096);
    float* pt_ts   = allocf((size_t)B * 4096);
    float* tr_tmp  = allocf((size_t)B * 4096);
    float* tt_tmp  = allocf((size_t)B * 4096);
    float* G0 = allocf((size_t)B * 1024);
    float* G1 = allocf((size_t)B * 1024);
    float* G2 = allocf((size_t)B * 1024);
    float* G3 = allocf((size_t)B * 1024);
    float* Gm  = allocf((size_t)B * 128);
    float* Gm2 = allocf((size_t)B * 128);
    float* vtb = allocf((size_t)B * N * 3);
    float* xb3 = allocf((size_t)B * N * 3);
    float* XH  = allocf((size_t)B * 64 * N);
    float* F0 = allocf((size_t)B * N * 128);
    float* F1 = allocf((size_t)B * N * 128);
    float* F2 = allocf((size_t)B * N * 128);
    float* F3 = allocf((size_t)B * N * 128);
    float* F4 = allocf((size_t)B * N * 128);
    float* F5 = allocf((size_t)B * N * 128);
    float* T0 = allocf((size_t)B * N * 1024);
    float* T1 = allocf((size_t)B * N * 1024);
    int* idx_x     = alloci((size_t)B * N * KNB);
    int* idx_v_pcl = alloci((size_t)B * N * KNB);
    int* idx_v_pr  = alloci((size_t)B * N * KNB);

    // ---- launch helpers ----
    auto gemm = [&](const float* A, const float* Bm, float* C, const float* bias,
                    const float* res, int M, int Nn, int K,
                    long sAb, long sAm, long sAk,
                    long sBb, long sBn, long sBk,
                    long sCb, long ldc, int act, int batch) {
        dim3 g((M + 15) / 16, (Nn + 31) / 32, batch);
        const bool ac = (sAk == 1);
        const bool bc = (sBk == 1);
        if (ac && bc)
            k_gemm_wmma<true, true><<<g, 32, 0, stream>>>(A, Bm, C, bias, res, M, Nn, K,
                sAb, sAm, sAk, sBb, sBn, sBk, sCb, ldc, act);
        else if (ac)
            k_gemm_wmma<true, false><<<g, 32, 0, stream>>>(A, Bm, C, bias, res, M, Nn, K,
                sAb, sAm, sAk, sBb, sBn, sBk, sCb, ldc, act);
        else if (bc)
            k_gemm_wmma<false, true><<<g, 32, 0, stream>>>(A, Bm, C, bias, res, M, Nn, K,
                sAb, sAm, sAk, sBb, sBn, sBk, sCb, ldc, act);
        else
            k_gemm_wmma<false, false><<<g, 32, 0, stream>>>(A, Bm, C, bias, res, M, Nn, K,
                sAb, sAm, sAk, sBb, sBn, sBk, sCb, ldc, act);
    };
    // lin over contiguous (B,N,Cin): Y = X @ W.T + b
    auto lin = [&](const float* X, const float* W, const float* bias, float* Y,
                   int Cin, int Cout, int act) {
        gemm(X, W, Y, bias, nullptr, N, Cout, Cin,
             (long)N * Cin, Cin, 1,  0, Cin, 1,  (long)N * Cout, Cout, act, B);
    };
    // fc over (B, Cin) rows (batch=1, M=B)
    auto fc = [&](const float* X, const float* W, const float* bias, float* Y,
                  int Cin, int Cout, int act) {
        gemm(X, W, Y, bias, nullptr, B, Cout, Cin,
             0, Cin, 1,  0, Cin, 1,  0, Cout, act, 1);
    };
    // ORL: dst = concat([feat, f_glob]) @ W.T + b + feat
    auto orl = [&](float* feat, const int* idx, const float* w, const float* bias,
                   int C, float* dst, float* tmp_nb) {
        k_gather_max<<<dim3(1, N, B), C, 0, stream>>>(feat, idx, tmp_nb, N, C);
        k_mean_n<<<(B * C + 255) / 256, 256, 0, stream>>>(tmp_nb, Gm, N, C);
        gemm(feat, w, dst, bias, feat, N, C, C,
             (long)N * C, C, 1,  0, 2L * C, 1,  (long)N * C, C, 0, B);
        gemm(Gm, w + C, Gm2, nullptr, nullptr, B, C, C,
             0, C, 1,  0, 2L * C, 1,  0, C, 0, 1);
        k_bcast_add<<<(int)(((long)B * N * C + 255) / 256), 256, 0, stream>>>(
            dst, Gm2, (long)B * N * C, N, C);
    };
    // mat_transform: o(B,64,N):  o[d,n] = sum_c x[c,n] * t[c,d]
    auto mtf = [&](const float* t, const float* x, float* o) {
        gemm(t, x, o, nullptr, nullptr, 64, N, 64,
             4096, 1, 64,  64L * N, 1, N,  64L * N, N, 0, B);
    };

    auto extract = [&](const float* v, float* xr, float* xts,
                       float* t_r, float* t_ts, int* idx_v) {
        // ---- point matrix ----
        gemm(v, pm_c1_w, F0, pm_c1_b, nullptr, N, 64, 3,
             3L * N, 1, N,  0, 3, 1,  (long)N * 64, 64, 1, B);
        gemm(F0, pm_c2_w, F1, pm_c2_b, nullptr, N, 128, 64,
             (long)N * 64, 64, 1,  0, 64, 1,  (long)N * 128, 128, 1, B);
        lin(F1, pm_c3_w, pm_c3_b, T0, 128, 1024, 1);
        k_rowmax<<<(B * 1024 + 255) / 256, 256, 0, stream>>>(T0, G0, N, 1024);
        fc(G0, pm_fc1_w, pm_fc1_b, G1, 1024, 512, 1);
        fc(G1, pm_fc2_w, pm_fc2_b, G2, 512, 256, 1);
        fc(G2, pm_fc3_w, pm_fc3_b, G3, 256, 9, 0);
        k_add_eye<<<1, 256, 0, stream>>>(G3, 3);
        // x = einsum('bcn,bcd->bnd', v, trans)
        gemm(v, G3, xb3, nullptr, nullptr, N, 3, 3,
             3L * N, 1, N,  9, 1, 3,  3L * N, 3, 0, B);
        // ---- hs_surface ----
        k_knn<<<dim3((N + 255) / 256, B), 256, 0, stream>>>(xb3, idx_x, N);
        gemm(xb3, hs1_ste_w, F0, hs1_ste_b, nullptr, N, 64, 3,
             3L * N, 3, 1,  0, 3, 1,  (long)N * 64, 64, 0, B);       // f_ste
        k_hs_feat<<<dim3(1, N, B), 64, 0, stream>>>(xb3, idx_x, dcn_hs1, nullptr, F1, N, 64);
        orl(F1, idx_x, hs1_orl_w, hs1_orl_b, 64, F2, F3);
        k_add_tr_relu<<<(int)(((long)B * N * 64 + 255) / 256), 256, 0, stream>>>(
            XH, F2, F0, N, 64, (long)B * N * 64);                    // XH (B,64,N)
        // ---- feat matrix ----
        k_tr3<<<(int)(((long)B * N * 3 + 255) / 256), 256, 0, stream>>>(vtb, v, N, (long)B * N * 3);
        k_knn<<<dim3((N + 255) / 256, B), 256, 0, stream>>>(vtb, idx_v, N);
        // fm1: fmap = XH^T (strided)
        gemm(XH, fm1_ste_w, F0, fm1_ste_b, nullptr, N, 64, 64,
             64L * N, 1, N,  0, 64, 1,  (long)N * 64, 64, 0, B);     // f_ste
        gemm(XH, fm1_W, T0, fm1_b, nullptr, N, 512, 64,
             64L * N, 1, N,  0, 1, 512,  (long)N * 512, 512, 0, B);  // fout (x@W)
        k_hs_feat<<<dim3(1, N, B), 64, 0, stream>>>(vtb, idx_v, dcn_fm1, T0, F1, N, 64);
        orl(F1, idx_v, fm1_orl_w, fm1_orl_b, 64, F2, F3);
        k_add<<<(int)(((long)B * N * 64 + 255) / 256), 256, 0, stream>>>(
            F4, F2, F0, (long)B * N * 64, 1);                        // h1 (B,N,64)
        // fm2
        gemm(F4, fm2_ste_w, F0, fm2_ste_b, nullptr, N, 128, 64,
             (long)N * 64, 64, 1,  0, 64, 1,  (long)N * 128, 128, 0, B);
        gemm(F4, fm2_W, T0, fm2_b, nullptr, N, 1024, 64,
             (long)N * 64, 64, 1,  0, 1, 1024,  (long)N * 1024, 1024, 0, B);
        k_hs_feat<<<dim3(1, N, B), 128, 0, stream>>>(vtb, idx_v, dcn_fm2, T0, F1, N, 128);
        orl(F1, idx_v, fm2_orl_w, fm2_orl_b, 128, F2, F3);
        k_add<<<(int)(((long)B * N * 128 + 255) / 256), 256, 0, stream>>>(
            F5, F2, F0, (long)B * N * 128, 1);                       // h2 (B,N,128)
        lin(F5, fm_c3_w, fm_c3_b, T0, 128, 1024, 1);
        k_rowmax<<<(B * 1024 + 255) / 256, 256, 0, stream>>>(T0, G0, N, 1024);
        fc(G0, fm_fc1_w, fm_fc1_b, G1, 1024, 512, 1);
        fc(G1, fm_fc2_w, fm_fc2_b, G2, 512, 256, 1);                 // g_pre
        fc(G2, fm_fc3_w, fm_fc3_b, t_r, 256, 4096, 0);
        k_add_eye<<<(B * 64 + 255) / 256, 256, 0, stream>>>(t_r, 64);
        fc(G2, fm_fc4_w, fm_fc4_b, t_ts, 256, 4096, 0);
        k_add_eye<<<(B * 64 + 255) / 256, 256, 0, stream>>>(t_ts, 64);
        // x_ts = T_ts(x); x_r = T_r(x_ts)
        mtf(t_ts, XH, xts);
        mtf(t_r, xts, xr);
    };

    auto dense = [&](const float* v, const float* xr, const float* xts,
                     const int* idx_v, float* out_r, float* out_ts) {
        k_tr3<<<(int)(((long)B * N * 3 + 255) / 256), 256, 0, stream>>>(vtb, v, N, (long)B * N * 3);
        k_sum_tr<<<(int)(((long)B * N * 64 + 255) / 256), 256, 0, stream>>>(
            F0, xr, xts, N, 64, (long)B * N * 64);                   // x (B,N,64)
        gemm(F0, hs2_ste_w, F1, hs2_ste_b, nullptr, N, 128, 64,
             (long)N * 64, 64, 1,  0, 64, 1,  (long)N * 128, 128, 0, B);
        gemm(F0, hs2_W, T0, hs2_b, nullptr, N, 1024, 64,
             (long)N * 64, 64, 1,  0, 1, 1024,  (long)N * 1024, 1024, 0, B);
        k_hs_feat<<<dim3(1, N, B), 128, 0, stream>>>(vtb, idx_v, dcn_hs2, T0, F2, N, 128);
        orl(F2, idx_v, hs2_orl_w, hs2_orl_b, 128, F3, F4);
        k_add<<<(int)(((long)B * N * 128 + 255) / 256), 256, 0, stream>>>(
            F5, F3, F1, (long)B * N * 128, 1);
        lin(F5, c3_w, c3_b, T1, 128, 512, 1);
        lin(T1, c4_w, c4_b, T0, 512, 1024, 0);
        k_rowmax<<<(B * 1024 + 255) / 256, 256, 0, stream>>>(T0, G0, N, 1024);
        const long tot = (long)B * 1088 * N;
        k_concat<<<(int)((tot + 255) / 256), 256, 0, stream>>>(out_r, G0, xr, N, tot);
        k_concat<<<(int)((tot + 255) / 256), 256, 0, stream>>>(out_ts, G0, xts, N, tot);
    };

    // ---- forward ----
    k_colnorm<<<(448 + 255) / 256, 256, 0, stream>>>(hs1_dirs, dcn_hs1, 448);
    k_colnorm<<<(448 + 255) / 256, 256, 0, stream>>>(fm1_dirs, dcn_fm1, 448);
    k_colnorm<<<(896 + 255) / 256, 256, 0, stream>>>(fm2_dirs, dcn_fm2, 896);
    k_colnorm<<<(896 + 255) / 256, 256, 0, stream>>>(hs2_dirs, dcn_hs2, 896);

    extract(v_pcl,   xr_pcl, xts_pcl, tr_tmp, tt_tmp, idx_v_pcl);
    extract(v_prior, xr_p,   xts_p,   pt_r,   pt_ts,  idx_v_pr);

    mtf(pt_r,  xr_pcl,  xr2);
    mtf(pt_ts, xts_pcl, xts2);

    float* out = (float*)d_out;
    const long seg = (long)B * 1088 * N;
    dense(v_pcl,   xr2,  xts2,  idx_v_pcl, out,           out + seg);
    dense(v_prior, xr_p, xts_p, idx_v_pr,  out + 2 * seg, out + 3 * seg);
}
